// STSProbSparseSelfAttention_45561013076445
// MI455X (gfx1250) — matrix operand order repaired and checked by reading, not compiled, not run on hardware
//
#include <hip/hip_runtime.h>
#include <hip/hip_bf16.h>
#include <math.h>

typedef __bf16 bf16;
typedef __attribute__((ext_vector_type(8)))  bf16  v8bf;
typedef __attribute__((ext_vector_type(16))) bf16  v16bf;
typedef __attribute__((ext_vector_type(8)))  float v8f;

#define L_    2048
#define B_    2
#define C_    512
#define H_    8
#define DH_   64
#define SK_   40
#define NTOP_ 40
#define BH_   16          // B_*H_
#define MPAD_ 64          // NTOP_ padded to multiple of 32 (2x2 wave tiling)
#define SCALING_ 0.125f   // DH^-0.5

// ---------------------------------------------------------------------------
// WMMA fragment loaders (contiguous k order, same for A and B, wave32)
//  A (16xK, row-major, lda elems): lane lo = M row; hi half selects K-subchunk
//    frag e[0..15] = A[row][k0 + 16*hi + 0..15]
//  B stored as Bt[N][K] (row-major, ldb elems): lane lo = N col
//    frag e[0..15] = Bt[col][k0 + 16*hi + 0..15]
// ---------------------------------------------------------------------------
__device__ __forceinline__ v16bf load_a_frag(const bf16* __restrict__ A, int lda,
                                             int row, int k0, int hi) {
  const bf16* p = A + (size_t)row * lda + k0 + hi * 16;
  v8bf c0 = *(const v8bf*)(p);
  v8bf c1 = *(const v8bf*)(p + 8);
  v16bf r;
#pragma unroll
  for (int i = 0; i < 8; ++i) { r[i] = c0[i]; r[i + 8] = c1[i]; }
  return r;
}

__device__ __forceinline__ v16bf load_b_frag(const bf16* __restrict__ Bt, int ldb,
                                             int col, int k0, int hi) {
  const bf16* p = Bt + (size_t)col * ldb + k0 + hi * 16;
  v8bf c0 = *(const v8bf*)(p);
  v8bf c1 = *(const v8bf*)(p + 8);
  v16bf r;
#pragma unroll
  for (int i = 0; i < 8; ++i) { r[i] = c0[i]; r[i + 8] = c1[i]; }
  return r;
}

// 2x2 register-tiled wave GEMM: 32x32 output per wave, 4 WMMAs per K-step.
__device__ __forceinline__ void wmma_tile_2x2(const bf16* __restrict__ A,
                                              const bf16* __restrict__ Bt,
                                              int lda, int ldb, int K,
                                              int m0, int n0, int lane,
                                              v8f acc[2][2]) {
  const int lo = lane & 15, hi = lane >> 4;
  for (int k0 = 0; k0 < K; k0 += 32) {
    __builtin_prefetch(A + (size_t)(m0 + lo) * lda + k0 + 64, 0, 1);
    __builtin_prefetch(Bt + (size_t)(n0 + lo) * ldb + k0 + 64, 0, 1);
    v16bf a0 = load_a_frag(A, lda, m0 + lo, k0, hi);
    v16bf a1 = load_a_frag(A, lda, m0 + 16 + lo, k0, hi);
    v16bf b0 = load_b_frag(Bt, ldb, n0 + lo, k0, hi);
    v16bf b1 = load_b_frag(Bt, ldb, n0 + 16 + lo, k0, hi);
    acc[0][0] = __builtin_amdgcn_wmma_f32_16x16x32_bf16(false, a0, false, b0,
                                                        (short)0, acc[0][0], false, false);
    acc[0][1] = __builtin_amdgcn_wmma_f32_16x16x32_bf16(false, a0, false, b1,
                                                        (short)0, acc[0][1], false, false);
    acc[1][0] = __builtin_amdgcn_wmma_f32_16x16x32_bf16(false, a1, false, b0,
                                                        (short)0, acc[1][0], false, false);
    acc[1][1] = __builtin_amdgcn_wmma_f32_16x16x32_bf16(false, a1, false, b1,
                                                        (short)0, acc[1][1], false, false);
  }
}

// ---------------------------------------------------------------------------
// f32 -> bf16 conversion
// ---------------------------------------------------------------------------
__global__ void cvt_bf16_kernel(const float* __restrict__ src,
                                bf16* __restrict__ dst, int n) {
  int i = blockIdx.x * blockDim.x + threadIdx.x;
  if (i < n) dst[i] = (bf16)src[i];
}

// ---------------------------------------------------------------------------
// Projection GEMM: y[m, j] = sum_k x[m,k] * W[j,k] ; m = l*B + b, j = h*64 + d
// Epilogue: (acc + bias[j]) * scale  ->  outF[b,h,l,d]  (+ optional bf16 copies)
// grid (M/32, N/(32*blockDim.y)), block (32, wavesPerBlock)
// ---------------------------------------------------------------------------
__global__ void proj_gemm_kernel(const bf16* __restrict__ X,
                                 const bf16* __restrict__ Wb,
                                 const float* __restrict__ bias, float scale,
                                 float* __restrict__ outF,
                                 bf16* __restrict__ outB,
                                 bf16* __restrict__ outBT) {
  const int lane = threadIdx.x;
  const int m0 = blockIdx.x * 32;
  const int n0 = (blockIdx.y * blockDim.y + threadIdx.y) * 32;
  v8f acc[2][2] = {};
  wmma_tile_2x2(X, Wb, C_, C_, C_, m0, n0, lane, acc);

  const int lo = lane & 15, hi = lane >> 4;
#pragma unroll
  for (int tj = 0; tj < 2; ++tj) {
    const int j = n0 + tj * 16 + lo;
    const int h = j >> 6, d = j & 63;
    const float bj = bias[j];
#pragma unroll
    for (int ti = 0; ti < 2; ++ti) {
#pragma unroll
      for (int r = 0; r < 8; ++r) {
        int m = m0 + ti * 16 + r + 8 * hi;
        int l = m >> 1, b = m & 1;
        float v = (acc[ti][tj][r] + bj) * scale;
        size_t o = ((size_t)(b * H_ + h) * L_ + l) * DH_ + d;
        outF[o] = v;
        if (outB) outB[o] = (bf16)v;
        if (outBT) outBT[((size_t)(b * H_ + h) * DH_ + d) * L_ + l] = (bf16)v;
      }
    }
  }
}

// ---------------------------------------------------------------------------
// Batched row-major GEMM: C[z][m, n] = sum_k A[z][m,k] * Bt[z][n,k] (+ bias[n])
// grid (M/32, N/(32*blockDim.y), Z), block (32, wavesPerBlock)
// ---------------------------------------------------------------------------
__global__ void gemm_rm_kernel(const bf16* __restrict__ A,
                               const bf16* __restrict__ Bt,
                               const float* __restrict__ bias,
                               float* __restrict__ Cout,
                               int K, int lda, int ldb, int ldc,
                               size_t sA, size_t sB, size_t sC) {
  const int z = blockIdx.z;
  const bf16* Az = A + (size_t)z * sA;
  const bf16* Bz = Bt + (size_t)z * sB;
  float* Cz = Cout + (size_t)z * sC;

  const int lane = threadIdx.x;
  const int m0 = blockIdx.x * 32;
  const int n0 = (blockIdx.y * blockDim.y + threadIdx.y) * 32;
  v8f acc[2][2] = {};
  wmma_tile_2x2(Az, Bz, lda, ldb, K, m0, n0, lane, acc);

  const int lo = lane & 15, hi = lane >> 4;
#pragma unroll
  for (int tj = 0; tj < 2; ++tj) {
    const int n = n0 + tj * 16 + lo;
    const float bn = bias ? bias[n] : 0.0f;
#pragma unroll
    for (int ti = 0; ti < 2; ++ti) {
#pragma unroll
      for (int r = 0; r < 8; ++r) {
        int m = m0 + ti * 16 + r + 8 * hi;
        Cz[(size_t)m * ldc + n] = acc[ti][tj][r] + bn;
      }
    }
  }
}

// ---------------------------------------------------------------------------
// sparsity[b,h,l] = max_s(q . k[idx_s]) - mean_s(q . k[idx_s])
// ---------------------------------------------------------------------------
__global__ void sparsity_kernel(const float* __restrict__ qf,
                                const float* __restrict__ kf,
                                const int* __restrict__ samp,
                                float* __restrict__ sp) {
  int t = blockIdx.x * blockDim.x + threadIdx.x;
  if (t >= BH_ * L_) return;
  int l = t & (L_ - 1);
  int bh = t >> 11;
  const float* qrow = qf + ((size_t)bh * L_ + l) * DH_;
  float qreg[DH_];
#pragma unroll
  for (int d = 0; d < DH_; ++d) qreg[d] = qrow[d];
  float mx = -INFINITY, sum = 0.0f;
  for (int s = 0; s < SK_; ++s) {
    int idx = samp[l * SK_ + s];
    const float* krow = kf + ((size_t)bh * L_ + idx) * DH_;
    float dot = 0.0f;
#pragma unroll
    for (int d = 0; d < DH_; ++d) dot = fmaf(qreg[d], krow[d], dot);
    mx = fmaxf(mx, dot);
    sum += dot;
  }
  sp[t] = mx - sum * (1.0f / SK_);
}

// ---------------------------------------------------------------------------
// top-40 per (b,h): iterative argmax over LDS copy (ties -> lowest index)
// grid (BH_), block (256)
// ---------------------------------------------------------------------------
__global__ void topk_kernel(const float* __restrict__ sp, int* __restrict__ topIdx) {
  __shared__ float vals[L_];
  __shared__ float rv[256];
  __shared__ int ri[256];
  const int bh = blockIdx.x, tid = threadIdx.x;
  for (int i = tid; i < L_; i += 256) vals[i] = sp[(size_t)bh * L_ + i];
  __syncthreads();
  for (int t = 0; t < NTOP_; ++t) {
    float bv = -INFINITY;
    int bi = 0x7fffffff;
    for (int i = tid; i < L_; i += 256) {
      float v = vals[i];
      if (v > bv) { bv = v; bi = i; }
    }
    rv[tid] = bv; ri[tid] = bi;
    __syncthreads();
    for (int s = 128; s > 0; s >>= 1) {
      if (tid < s) {
        if (rv[tid + s] > rv[tid] ||
            (rv[tid + s] == rv[tid] && ri[tid + s] < ri[tid])) {
          rv[tid] = rv[tid + s]; ri[tid] = ri[tid + s];
        }
      }
      __syncthreads();
    }
    if (tid == 0) { topIdx[bh * NTOP_ + t] = ri[0]; vals[ri[0]] = -INFINITY; }
    __syncthreads();
  }
}

// ---------------------------------------------------------------------------
// Qr[bh][n][d] = bf16(q[bh][topIdx[n]][d]) ; rows 40..63 zero
// ---------------------------------------------------------------------------
__global__ void build_qr_kernel(const float* __restrict__ qf,
                                const int* __restrict__ topIdx,
                                bf16* __restrict__ Qr) {
  int t = blockIdx.x * blockDim.x + threadIdx.x;
  if (t >= BH_ * MPAD_ * DH_) return;
  int d = t & 63;
  int n = (t >> 6) % MPAD_;
  int bh = t / (MPAD_ * DH_);
  float v = 0.0f;
  if (n < NTOP_) {
    int l = topIdx[bh * NTOP_ + n];
    v = qf[((size_t)bh * L_ + l) * DH_ + d];
  }
  Qr[t] = (bf16)v;
}

// ---------------------------------------------------------------------------
// masked softmax: mask(l') = (l'/32 <= p/32)  => valid prefix [0, 32*(p/32+1))
// one wave per row; attn written as bf16 (pad rows zeroed)
// ---------------------------------------------------------------------------
__global__ void softmax_kernel(const float* __restrict__ scores,
                               const int* __restrict__ topIdx,
                               bf16* __restrict__ attn) {
  int gw = (blockIdx.x * blockDim.x + threadIdx.x) >> 5;
  int lane = threadIdx.x & 31;
  if (gw >= BH_ * MPAD_) return;
  int n = gw % MPAD_;
  int bh = gw / MPAD_;
  bf16* arow = attn + (size_t)gw * L_;
  if (n >= NTOP_) {
    for (int c = lane; c < L_; c += 32) arow[c] = (bf16)0.0f;
    return;
  }
  int p = topIdx[bh * NTOP_ + n];
  int lvalid = ((p >> 5) + 1) << 5;
  const float* srow = scores + (size_t)gw * L_;
  float mx = -INFINITY;
  for (int c = lane; c < lvalid; c += 32) mx = fmaxf(mx, srow[c]);
#pragma unroll
  for (int o = 16; o > 0; o >>= 1) mx = fmaxf(mx, __shfl_xor(mx, o, 32));
  float sum = 0.0f;
  for (int c = lane; c < lvalid; c += 32) sum += __expf(srow[c] - mx);
#pragma unroll
  for (int o = 16; o > 0; o >>= 1) sum += __shfl_xor(sum, o, 32);
  float inv = 1.0f / sum;
  for (int c = lane; c < L_; c += 32) {
    float a = (c < lvalid) ? __expf(srow[c] - mx) * inv : 0.0f;
    arow[c] = (bf16)a;
  }
}

// ---------------------------------------------------------------------------
// ctx[b][l][h*64+d] = bf16(mean_l v[b,h,l,d]) ; grid(BH_), block(DH_)
// ---------------------------------------------------------------------------
__global__ void ctx_fill_kernel(const float* __restrict__ vf, bf16* __restrict__ ctx) {
  int bh = blockIdx.x, d = threadIdx.x;
  int b = bh >> 3, h = bh & 7;
  const float* vb = vf + (size_t)bh * L_ * DH_;
  float s = 0.0f;
  for (int l = 0; l < L_; ++l) s += vb[(size_t)l * DH_ + d];
  bf16 m = (bf16)(s * (1.0f / (float)L_));
  bf16* base = ctx + (size_t)b * L_ * C_ + h * DH_ + d;
  for (int l = 0; l < L_; ++l) base[(size_t)l * C_] = m;
}

// ---------------------------------------------------------------------------
// scatter: ctx[b][topIdx[n]][h*64+d] = bf16(updates[bh][n][d])
// ---------------------------------------------------------------------------
__global__ void ctx_scatter_kernel(const float* __restrict__ updates,
                                   const int* __restrict__ topIdx,
                                   bf16* __restrict__ ctx) {
  int t = blockIdx.x * blockDim.x + threadIdx.x;
  if (t >= BH_ * NTOP_ * DH_) return;
  int d = t & 63;
  int n = (t >> 6) % NTOP_;
  int bh = t / (NTOP_ * DH_);
  int b = bh >> 3, h = bh & 7;
  int l = topIdx[bh * NTOP_ + n];
  ctx[((size_t)b * L_ + l) * C_ + h * DH_ + d] =
      (bf16)updates[((size_t)bh * MPAD_ + n) * DH_ + d];
}

// ---------------------------------------------------------------------------
extern "C" void kernel_launch(void* const* d_in, const int* in_sizes, int n_in,
                              void* d_out, int out_size, void* d_ws, size_t ws_size,
                              hipStream_t stream) {
  const float* query = (const float*)d_in[0];
  const float* key   = (const float*)d_in[1];
  const float* value = (const float*)d_in[2];
  const int*   samp  = (const int*)d_in[3];
  const float* Wq = (const float*)d_in[4];  const float* bq = (const float*)d_in[5];
  const float* Wk = (const float*)d_in[6];  const float* bk = (const float*)d_in[7];
  const float* Wv = (const float*)d_in[8];  const float* bv = (const float*)d_in[9];
  const float* Wo = (const float*)d_in[10]; const float* bo = (const float*)d_in[11];
  float* out = (float*)d_out;

  // workspace carve-up (256B aligned)
  char* w = (char*)d_ws;
  size_t off = 0;
  auto alloc = [&](size_t bytes) -> void* {
    void* p = w + off;
    off = (off + bytes + 255) & ~(size_t)255;
    return p;
  };
  const size_t NX = (size_t)L_ * B_ * C_;     // 2,097,152
  const size_t NW = (size_t)C_ * C_;          //   262,144
  const size_t NQ = (size_t)BH_ * L_ * DH_;   // 2,097,152

  bf16* xq_b = (bf16*)alloc(NX * 2);
  bf16* xk_b = (bf16*)alloc(NX * 2);
  bf16* xv_b = (bf16*)alloc(NX * 2);
  bf16* Wq_b = (bf16*)alloc(NW * 2);
  bf16* Wk_b = (bf16*)alloc(NW * 2);
  bf16* Wv_b = (bf16*)alloc(NW * 2);
  bf16* Wo_b = (bf16*)alloc(NW * 2);
  float* qf  = (float*)alloc(NQ * 4);
  float* kf  = (float*)alloc(NQ * 4);
  float* vf  = (float*)alloc(NQ * 4);
  bf16* kb   = (bf16*)alloc(NQ * 2);          // [B,H,L,DH] bf16
  bf16* vT   = (bf16*)alloc(NQ * 2);          // [B,H,DH,L] bf16
  float* sp  = (float*)alloc((size_t)BH_ * L_ * 4);
  int* topIdx = (int*)alloc((size_t)BH_ * NTOP_ * 4);
  bf16* Qr   = (bf16*)alloc((size_t)BH_ * MPAD_ * DH_ * 2);
  float* scores = (float*)alloc((size_t)BH_ * MPAD_ * L_ * 4);
  bf16* attn = (bf16*)alloc((size_t)BH_ * MPAD_ * L_ * 2);
  float* updates = (float*)alloc((size_t)BH_ * MPAD_ * DH_ * 4);
  bf16* ctx  = (bf16*)alloc(NX * 2);          // [B,L,C] bf16
  (void)ws_size; (void)n_in; (void)in_sizes; (void)out_size;

  // 1) bf16 conversions
  cvt_bf16_kernel<<<(int)((NX + 255) / 256), 256, 0, stream>>>(query, xq_b, (int)NX);
  cvt_bf16_kernel<<<(int)((NX + 255) / 256), 256, 0, stream>>>(key,   xk_b, (int)NX);
  cvt_bf16_kernel<<<(int)((NX + 255) / 256), 256, 0, stream>>>(value, xv_b, (int)NX);
  cvt_bf16_kernel<<<(int)((NW + 255) / 256), 256, 0, stream>>>(Wq, Wq_b, (int)NW);
  cvt_bf16_kernel<<<(int)((NW + 255) / 256), 256, 0, stream>>>(Wk, Wk_b, (int)NW);
  cvt_bf16_kernel<<<(int)((NW + 255) / 256), 256, 0, stream>>>(Wv, Wv_b, (int)NW);
  cvt_bf16_kernel<<<(int)((NW + 255) / 256), 256, 0, stream>>>(Wo, Wo_b, (int)NW);

  // 2) WMMA projections: M=4096, N=512, K=512 (32x32 per wave, 128 cols/block)
  dim3 pgrid(L_ * B_ / 32, C_ / 128, 1), pblk(32, 4, 1);
  proj_gemm_kernel<<<pgrid, pblk, 0, stream>>>(xq_b, Wq_b, bq, SCALING_, qf, nullptr, nullptr);
  proj_gemm_kernel<<<pgrid, pblk, 0, stream>>>(xk_b, Wk_b, bk, 1.0f, kf, kb, nullptr);
  proj_gemm_kernel<<<pgrid, pblk, 0, stream>>>(xv_b, Wv_b, bv, 1.0f, vf, nullptr, vT);

  // 3) sampled-score sparsity measure
  sparsity_kernel<<<(BH_ * L_ + 255) / 256, 256, 0, stream>>>(qf, kf, samp, sp);

  // 4) top-40 per (b,h)
  topk_kernel<<<BH_, 256, 0, stream>>>(sp, topIdx);

  // 5) gather Qr (bf16, padded to 64 rows)
  build_qr_kernel<<<(BH_ * MPAD_ * DH_ + 255) / 256, 256, 0, stream>>>(qf, topIdx, Qr);

  // 6) scores = Qr @ k^T : batched WMMA, M=64, N=2048, K=64
  gemm_rm_kernel<<<dim3(MPAD_ / 32, L_ / 128, BH_), pblk, 0, stream>>>(
      Qr, kb, nullptr, scores, DH_, DH_, DH_, L_,
      (size_t)MPAD_ * DH_, (size_t)L_ * DH_, (size_t)MPAD_ * L_);

  // 7) block-causal-masked softmax -> bf16 attn
  softmax_kernel<<<(BH_ * MPAD_ * 32 + 255) / 256, 256, 0, stream>>>(scores, topIdx, attn);

  // 8) updates = attn @ v : batched WMMA, M=64, N=64, K=2048 (2 waves/block)
  gemm_rm_kernel<<<dim3(MPAD_ / 32, 1, BH_), dim3(32, 2, 1), 0, stream>>>(
      attn, vT, nullptr, updates, L_, L_, L_, DH_,
      (size_t)MPAD_ * L_, (size_t)DH_ * L_, (size_t)MPAD_ * DH_);

  // 9) ctx = broadcast mean of v, then scatter updates at top indices
  ctx_fill_kernel<<<BH_, DH_, 0, stream>>>(vf, ctx);
  ctx_scatter_kernel<<<(BH_ * NTOP_ * DH_ + 255) / 256, 256, 0, stream>>>(updates, topIdx, ctx);

  // 10) out = ctx @ Wo^T + bo : WMMA, M=4096, N=512, K=512, row-major out
  gemm_rm_kernel<<<dim3(B_ * L_ / 32, C_ / 128, 1), pblk, 0, stream>>>(
      ctx, Wo_b, bo, out, C_, C_, C_, C_, 0, 0, 0);
}
